// COM_NET_30932354465909
// MI455X (gfx1250) — matrix operand order repaired and checked by reading, not compiled, not run on hardware
//
#include <hip/hip_runtime.h>
#include <hip/hip_bf16.h>
#include <cstdint>
#include <cstddef>

// ---------------------------------------------------------------------------
// Problem constants (match reference)
// ---------------------------------------------------------------------------
#define GN  50000
#define GE  800000
#define IN1 512
#define HID1 256
#define OUT1 64
#define INS 256
#define HIDS 128
#define OUTS 64
#define NEG_SLOPE 0.2f

typedef __attribute__((ext_vector_type(16))) __bf16 v16bf;
typedef __attribute__((ext_vector_type(8)))  float  v8f;
typedef unsigned int v4u __attribute__((ext_vector_type(4)));
typedef int          v8i __attribute__((ext_vector_type(8)));
typedef int          v4i __attribute__((ext_vector_type(4)));

union AFrag { v16bf v; unsigned int w[8]; uint4 q[2]; };

// ---------------------------------------------------------------------------
// Helpers
// ---------------------------------------------------------------------------
static __device__ __forceinline__ unsigned short f32_to_bf16_rne(float f) {
  unsigned u = __float_as_uint(f);
  u += 0x7FFFu + ((u >> 16) & 1u);
  return (unsigned short)(u >> 16);
}
// Monotone order-preserving map float -> uint (for atomicMax-based segment max)
static __device__ __forceinline__ unsigned enc_f(float f) {
  unsigned u = __float_as_uint(f);
  return (u & 0x80000000u) ? ~u : (u | 0x80000000u);
}
static __device__ __forceinline__ float dec_f(unsigned u) {
  return (u & 0x80000000u) ? __uint_as_float(u & 0x7FFFFFFFu)
                           : __uint_as_float(~u);
}

// ---------------------------------------------------------------------------
// Elementwise kernels
// ---------------------------------------------------------------------------
__global__ void __launch_bounds__(256)
k_fill_u32(unsigned* __restrict__ p, unsigned val, long long n) {
  long long i = (long long)blockIdx.x * blockDim.x + threadIdx.x;
  if (i < n) p[i] = val;
}

__global__ void __launch_bounds__(256)
k_cvt_bf16(const float* __restrict__ in, unsigned short* __restrict__ out, long long n) {
  long long i = (long long)blockIdx.x * blockDim.x + threadIdx.x;
  if (i < n) out[i] = f32_to_bf16_rne(in[i]);
}

// out(C x R) = transpose of in(R x C), converted to bf16 (for tied weights)
__global__ void __launch_bounds__(256)
k_transpose_cvt_bf16(const float* __restrict__ in, unsigned short* __restrict__ out,
                     int R, int C) {
  int i = blockIdx.x * blockDim.x + threadIdx.x;
  if (i >= R * C) return;
  int r = i / C, c = i % C;
  out[(long long)c * R + r] = f32_to_bf16_rne(in[i]);
}

__global__ void __launch_bounds__(256)
k_elu_to_bf16(const float* __restrict__ in, unsigned short* __restrict__ out, long long n) {
  long long i = (long long)blockIdx.x * blockDim.x + threadIdx.x;
  if (i >= n) return;
  float x = in[i];
  float y = x > 0.0f ? x : (__expf(x) - 1.0f);
  out[i] = f32_to_bf16_rne(y);
}

// ---------------------------------------------------------------------------
// WMMA GEMM with TDM-staged weights:
//   C(M x Nc) = A(M x K) * B(Nc x K)^T      (bf16 in, fp32 out)
// Block = 8 waves = 128 rows x 64 cols. The 64xK B tile is DMA'd into LDS
// once per block by the Tensor Data Mover (padded rows: K+8 ushorts to avoid
// LDS bank conflicts), then each wave runs the K loop: A from global
// (global_load_b128), B fragments from LDS (ds_load_b128), 4 WMMA
// accumulators (16x64 strip) per wave.
// Requires: M%16==0, Nc%64==0, K%32==0, K in {64,128,256,512}.
// ---------------------------------------------------------------------------
__global__ void __launch_bounds__(256)
k_gemm_bf16_nt(const unsigned short* __restrict__ A,
               const unsigned short* __restrict__ B,
               float* __restrict__ C,
               int M, int Nc, int K, int rowBlocks) {
  extern __shared__ unsigned short sB[];          // 64 x (K+8) bf16
  const int lane     = threadIdx.x & 31;
  const int waveId   = threadIdx.x >> 5;
  const int rowBlock = blockIdx.x % rowBlocks;
  const int colGroup = blockIdx.x / rowBlocks;
  const int colBase  = colGroup << 6;
  const int ldsStride = K + 8;                    // ushorts per staged row

#if __has_builtin(__builtin_amdgcn_tensor_load_to_lds)
  if (waveId == 0) {
    // ---- Tensor DMA descriptor (CDNA5 ISA ch.8): 64 x K bf16 tile ----
    unsigned long long gaddr =
        (unsigned long long)(const void*)(B + (size_t)colBase * K);
    unsigned ldsOff = (unsigned)(unsigned long long)(const void*)sB;
    int pi = (31 - __clz(K)) - 2;                 // pad_interval: one row = K/2 dwords
    v4u g0;
    g0[0] = 1u;                                   // count=1 (valid descriptor)
    g0[1] = ldsOff;                               // lds_addr (bytes)
    g0[2] = (unsigned)gaddr;                      // global_addr[31:0]
    g0[3] = ((unsigned)(gaddr >> 32) & 0x01FFFFFFu) | 0x80000000u;  // addr[56:32] | type=2
    v8i g1;
    g1[0] = (int)((1u << 16) | (1u << 20) |       // data_size=2B, pad_enable
                  ((unsigned)pi << 22) |          // pad_interval = one row
                  (3u << 25));                    // pad_amount = 4 dwords (16B)
    g1[1] = (int)((unsigned)K << 16);             // tensor_dim0 = K (low bits at 63:48)
    g1[2] = (int)(64u << 16);                     // tensor_dim1 = 64 (bits 95:80)
    g1[3] = (int)((unsigned)K << 16);             // tile_dim0 = K (bits 127:112)
    g1[4] = 64;                                   // tile_dim1 = 64 rows
    g1[5] = K;                                    // tensor_dim0_stride = K
    g1[6] = 0;
    g1[7] = 0;
    v4i gz  = {0, 0, 0, 0};                       // groups 2/3 unused (2-D tile)
    v8i gz8 = {0, 0, 0, 0, 0, 0, 0, 0};           // extension group (zero/NULL)
    __builtin_amdgcn_tensor_load_to_lds(g0, g1, gz, gz, gz8, 0);
    __builtin_amdgcn_s_wait_tensorcnt(0);
  }
#else
  // Fallback: cooperative staging (16B chunks) with the same padded layout.
  for (int idx = threadIdx.x; idx < 64 * (K >> 3); idx += blockDim.x) {
    int row = idx / (K >> 3);
    int c8  = (idx % (K >> 3)) << 3;
    *(uint4*)&sB[row * ldsStride + c8] =
        *(const uint4*)&B[(size_t)(colBase + row) * K + c8];
  }
#endif
  __syncthreads();

  const int rowTile = rowBlock * 8 + waveId;
  if (rowTile >= (M >> 4)) return;                // wave-uniform exit
  const int rowBase = rowTile << 4;
  const int m  = lane & 15;
  const int hi = lane >> 4;

  const unsigned short* aRow = A + (size_t)(rowBase + m) * K;

  v8f acc[4];
#pragma unroll
  for (int ct = 0; ct < 4; ++ct) acc[ct] = (v8f)0.0f;

  for (int k0 = 0; k0 < K; k0 += 32) {
    AFrag a;
    a.q[0] = *(const uint4*)(aRow + k0 + hi * 8);        // K = k0+hi*8 .. +7
    a.q[1] = *(const uint4*)(aRow + k0 + 16 + hi * 8);   // K = k0+16+hi*8 .. +7

#pragma unroll
    for (int ct = 0; ct < 4; ++ct) {
      const uint4* bp =
          (const uint4*)(sB + (ct * 16 + m) * ldsStride + k0 + hi * 8);
      AFrag b;
      b.q[0] = bp[0];                                    // ds_load_b128
      b.q[1] = bp[2];                                    // ds_load_b128
      acc[ct] = __builtin_amdgcn_wmma_f32_16x16x32_bf16(
          false, a.v, false, b.v, (short)0, acc[ct], false, false);
    }
  }

#pragma unroll
  for (int ct = 0; ct < 4; ++ct) {
#pragma unroll
    for (int r = 0; r < 8; ++r) {
      int row = rowBase + hi * 8 + r;
      int col = colBase + ct * 16 + m;
      C[(size_t)row * Nc + col] = acc[ct][r];
    }
  }
}

// ---------------------------------------------------------------------------
// Attention pieces
// ---------------------------------------------------------------------------
__global__ void __launch_bounds__(256)
k_rowdot2(const float* __restrict__ h, const float* __restrict__ a,
          const float* __restrict__ b, float* __restrict__ ss,
          float* __restrict__ sd, int n, int H) {
  int wave = blockIdx.x * (blockDim.x >> 5) + (threadIdx.x >> 5);
  int lane = threadIdx.x & 31;
  if (wave >= n) return;
  const float* row = h + (size_t)wave * H;
  float s0 = 0.0f, s1 = 0.0f;
  for (int k = lane; k < H; k += 32) {
    float v = row[k];
    s0 += v * a[k];
    s1 += v * b[k];
  }
  for (int off = 16; off > 0; off >>= 1) {
    s0 += __shfl_down(s0, off, 32);
    s1 += __shfl_down(s1, off, 32);
  }
  if (lane == 0) { ss[wave] = s0; sd[wave] = s1; }
}

__global__ void __launch_bounds__(256)
k_edge_logit_max(const float* __restrict__ ss, const float* __restrict__ sd,
                 const int* __restrict__ src, const int* __restrict__ dst,
                 float* __restrict__ logits, unsigned* __restrict__ mbits, int ne) {
  int e = blockIdx.x * blockDim.x + threadIdx.x;
  if (e >= ne) return;
  float x = ss[src[e]] + sd[dst[e]];
  float lo = x > 0.0f ? x : NEG_SLOPE * x;
  logits[e] = lo;
  atomicMax(&mbits[dst[e]], enc_f(lo));
}

__global__ void __launch_bounds__(256)
k_edge_exp_sum(float* __restrict__ logits, const int* __restrict__ dst,
               const unsigned* __restrict__ mbits, float* __restrict__ ssum, int ne) {
  int e = blockIdx.x * blockDim.x + threadIdx.x;
  if (e >= ne) return;
  int d = dst[e];
  float ex = __expf(logits[e] - dec_f(mbits[d]));
  logits[e] = ex;
  atomicAdd(&ssum[d], ex);
}

__global__ void __launch_bounds__(256)
k_edge_alpha(float* __restrict__ alpha, const int* __restrict__ dst,
             const float* __restrict__ ssum, int ne) {
  int e = blockIdx.x * blockDim.x + threadIdx.x;
  if (e >= ne) return;
  alpha[e] = alpha[e] / (ssum[dst[e]] + 1e-16f);
}

// agg[dst] += alpha_e * h[src]  (wave/edge, float4 gathers, hw f32 atomics)
// Requires H % 128 == 0 (H = 256 or 128 here).
__global__ void __launch_bounds__(256)
k_propagate(const float* __restrict__ h, const float* __restrict__ alpha,
            const int* __restrict__ src, const int* __restrict__ dst,
            float* __restrict__ agg, int ne, int H) {
  int wave = blockIdx.x * (blockDim.x >> 5) + (threadIdx.x >> 5);
  int lane = threadIdx.x & 31;
  if (wave >= ne) return;
  int s = src[wave], d = dst[wave];
  float al = alpha[wave];
  const float* hs = h + (size_t)s * H;
  float* ag = agg + (size_t)d * H;
  for (int k = lane * 4; k < H; k += 128) {
    float4 v = *(const float4*)(hs + k);
    atomicAdd(&ag[k + 0], al * v.x);
    atomicAdd(&ag[k + 1], al * v.y);
    atomicAdd(&ag[k + 2], al * v.z);
    atomicAdd(&ag[k + 3], al * v.w);
  }
}

// row L2-normalize, 64 cols (wave per node)
__global__ void __launch_bounds__(256)
k_normalize64(const float* __restrict__ in, float* __restrict__ out, int n) {
  int wave = blockIdx.x * (blockDim.x >> 5) + (threadIdx.x >> 5);
  int lane = threadIdx.x & 31;
  if (wave >= n) return;
  float v0 = in[(size_t)wave * 64 + lane];
  float v1 = in[(size_t)wave * 64 + 32 + lane];
  float s = v0 * v0 + v1 * v1;
  for (int off = 16; off > 0; off >>= 1) s += __shfl_down(s, off, 32);
  s = __shfl(s, 0, 32);
  float inv = 1.0f / (sqrtf(s) + 1e-12f);
  out[(size_t)wave * 64 + lane]      = v0 * inv;
  out[(size_t)wave * 64 + 32 + lane] = v1 * inv;
}

// ---------------------------------------------------------------------------
// Host-side helpers
// ---------------------------------------------------------------------------
static inline int blks(long long n, int t) { return (int)((n + t - 1) / t); }
static inline int wblks(long long waves)   { return (int)((waves + 7) / 8); }

static void gemm(const unsigned short* A, const unsigned short* B, float* C,
                 int M, int Nc, int K, hipStream_t s) {
  int rowBlocks = ((M >> 4) + 7) / 8;
  int colGroups = Nc >> 6;
  size_t smem = (size_t)64 * (K + 8) * 2;
  k_gemm_bf16_nt<<<rowBlocks * colGroups, 256, smem, s>>>(A, B, C, M, Nc, K, rowBlocks);
}

struct Ws {
  unsigned short *xb, *hb, *h2b, *w1b, *w2b, *w1tb, *w2tb;
  float *h, *agg, *ss, *sd, *ssum, *alpha, *h2f;
  unsigned *mbits;
};

extern "C" void kernel_launch(void* const* d_in, const int* in_sizes, int n_in,
                              void* d_out, int out_size, void* d_ws, size_t ws_size,
                              hipStream_t stream) {
  (void)in_sizes; (void)n_in; (void)out_size; (void)ws_size;

  // ---- workspace carve (deterministic, recomputed each call) ----
  char* base = (char*)d_ws;
  size_t off = 0;
  auto carve = [&](size_t bytes) -> char* {
    char* p = base + off;
    off = (off + bytes + 255) & ~(size_t)255;
    return p;
  };
  Ws ws;
  ws.xb    = (unsigned short*)carve((size_t)GN * IN1 * 2);
  ws.h     = (float*)carve((size_t)GN * HID1 * 4);
  ws.agg   = (float*)carve((size_t)GN * HID1 * 4);
  ws.hb    = (unsigned short*)carve((size_t)GN * HID1 * 2);
  ws.ss    = (float*)carve((size_t)GN * 4);
  ws.sd    = (float*)carve((size_t)GN * 4);
  ws.mbits = (unsigned*)carve((size_t)GN * 4);
  ws.ssum  = (float*)carve((size_t)GN * 4);
  ws.alpha = (float*)carve((size_t)GE * 4);
  ws.h2b   = (unsigned short*)carve((size_t)GN * OUT1 * 2);
  ws.h2f   = (float*)carve((size_t)GN * OUTS * 4);
  ws.w1b   = (unsigned short*)carve((size_t)HID1 * IN1 * 2);
  ws.w2b   = (unsigned short*)carve((size_t)OUT1 * HID1 * 2);
  ws.w1tb  = (unsigned short*)carve((size_t)IN1 * HID1 * 2);
  ws.w2tb  = (unsigned short*)carve((size_t)HID1 * OUT1 * 2);

  float* out = (float*)d_out;
  float* emb[2] = { out,                        out + (size_t)GN * OUT1 };
  float* de[2]  = { out + 2ll * GN * OUT1,      out + 2ll * GN * OUT1 + (size_t)GN * IN1 };
  float* sh[2]  = { out + 2ll * GN * OUT1 + 2ll * GN * IN1,
                    out + 2ll * GN * OUT1 + 2ll * GN * IN1 + (size_t)GN * OUTS };

  auto attention = [&](const float* hmat, const float* a_s, const float* a_d,
                       const int* src, const int* dst, int H) {
    k_rowdot2<<<wblks(GN), 256, 0, stream>>>(hmat, a_s, a_d, ws.ss, ws.sd, GN, H);
    k_fill_u32<<<blks(GN, 256), 256, 0, stream>>>(ws.mbits, 0u, GN);
    k_fill_u32<<<blks(GN, 256), 256, 0, stream>>>((unsigned*)ws.ssum, 0u, GN);
    k_edge_logit_max<<<blks(GE, 256), 256, 0, stream>>>(ws.ss, ws.sd, src, dst,
                                                        ws.alpha, ws.mbits, GE);
    k_edge_exp_sum<<<blks(GE, 256), 256, 0, stream>>>(ws.alpha, dst, ws.mbits,
                                                      ws.ssum, GE);
    k_edge_alpha<<<blks(GE, 256), 256, 0, stream>>>(ws.alpha, dst, ws.ssum, GE);
  };
  auto prop = [&](const float* hmat, const int* src, const int* dst, int H) {
    k_fill_u32<<<blks((long long)GN * H, 256), 256, 0, stream>>>(
        (unsigned*)ws.agg, 0u, (long long)GN * H);
    k_propagate<<<wblks(GE), 256, 0, stream>>>(hmat, ws.alpha, src, dst, ws.agg, GE, H);
  };

  // ================= omics GAT autoencoders =================
  for (int i = 0; i < 2; ++i) {
    const float* x   = (const float*)d_in[i];
    const int*   src = (const int*)d_in[4 + i];
    const int*   dst = src + GE;
    const float* W1  = (const float*)d_in[6 + 4 * i];   // (HID1, IN1)
    const float* a_s = (const float*)d_in[7 + 4 * i];
    const float* a_d = (const float*)d_in[8 + 4 * i];
    const float* W2  = (const float*)d_in[9 + 4 * i];   // (OUT1, HID1)

    k_cvt_bf16<<<blks((long long)HID1 * IN1, 256), 256, 0, stream>>>(W1, ws.w1b, (long long)HID1 * IN1);
    k_cvt_bf16<<<blks((long long)OUT1 * HID1, 256), 256, 0, stream>>>(W2, ws.w2b, (long long)OUT1 * HID1);
    k_transpose_cvt_bf16<<<blks((long long)OUT1 * HID1, 256), 256, 0, stream>>>(W2, ws.w2tb, OUT1, HID1);
    k_transpose_cvt_bf16<<<blks((long long)HID1 * IN1, 256), 256, 0, stream>>>(W1, ws.w1tb, HID1, IN1);
    k_cvt_bf16<<<blks((long long)GN * IN1, 256), 256, 0, stream>>>(x, ws.xb, (long long)GN * IN1);

    // conv1: h = x @ W1^T                               [N x 256]
    gemm(ws.xb, ws.w1b, ws.h, GN, HID1, IN1, stream);
    attention(ws.h, a_s, a_d, src, dst, HID1);
    prop(ws.h, src, dst, HID1);                          // h1 = propagate(h)
    k_elu_to_bf16<<<blks((long long)GN * HID1, 256), 256, 0, stream>>>(
        ws.agg, ws.hb, (long long)GN * HID1);

    // conv2: emb = h1 @ W2^T                            [N x 64]
    gemm(ws.hb, ws.w2b, emb[i], GN, OUT1, HID1, stream);

    // conv3: t = emb @ W2; h3 = elu(propagate(t))       [N x 256]
    k_cvt_bf16<<<blks((long long)GN * OUT1, 256), 256, 0, stream>>>(
        emb[i], ws.h2b, (long long)GN * OUT1);
    gemm(ws.h2b, ws.w2tb, ws.h, GN, HID1, OUT1, stream);
    prop(ws.h, src, dst, HID1);
    k_elu_to_bf16<<<blks((long long)GN * HID1, 256), 256, 0, stream>>>(
        ws.agg, ws.hb, (long long)GN * HID1);

    // conv4: de = h3 @ W1                               [N x 512]
    gemm(ws.hb, ws.w1tb, de[i], GN, IN1, HID1, stream);
  }

  // ================= shared encoder (same params, both graphs) =================
  const float* W1s  = (const float*)d_in[14];   // (HIDS, INS)
  const float* a_ss = (const float*)d_in[15];
  const float* a_ds = (const float*)d_in[16];
  const float* W2s  = (const float*)d_in[17];   // (OUTS, HIDS)
  k_cvt_bf16<<<blks((long long)HIDS * INS, 256), 256, 0, stream>>>(W1s, ws.w1b, (long long)HIDS * INS);
  k_cvt_bf16<<<blks((long long)OUTS * HIDS, 256), 256, 0, stream>>>(W2s, ws.w2b, (long long)OUTS * HIDS);

  for (int i = 0; i < 2; ++i) {
    const float* x   = (const float*)d_in[2 + i];
    const int*   src = (const int*)d_in[4 + i];
    const int*   dst = src + GE;

    k_cvt_bf16<<<blks((long long)GN * INS, 256), 256, 0, stream>>>(x, ws.xb, (long long)GN * INS);
    gemm(ws.xb, ws.w1b, ws.h, GN, HIDS, INS, stream);    // [N x 128]
    attention(ws.h, a_ss, a_ds, src, dst, HIDS);
    prop(ws.h, src, dst, HIDS);
    k_elu_to_bf16<<<blks((long long)GN * HIDS, 256), 256, 0, stream>>>(
        ws.agg, ws.hb, (long long)GN * HIDS);
    gemm(ws.hb, ws.w2b, ws.h2f, GN, OUTS, HIDS, stream); // [N x 64]
    k_normalize64<<<wblks(GN), 256, 0, stream>>>(ws.h2f, sh[i], GN);
  }
}
